// DeepseekMoE_35476429865908
// MI455X (gfx1250) — matrix operand order repaired and checked by reading, not compiled, not run on hardware
//
#include <hip/hip_runtime.h>
#include <stdint.h>

#define N_TOK 8192
#define DDIM  1024
#define HDIM  2048
#define ODIM  1024
#define NEXP  8
#define LDSS  72   // LDS row stride in bf16 elems for a 64-wide K tile (16B aligned, conflict-free)

typedef __attribute__((ext_vector_type(16))) __bf16 v16bf;
typedef __attribute__((ext_vector_type(8)))  float  v8f;

__device__ __forceinline__ unsigned pk2(float a, float b) {
  unsigned ua = __float_as_uint(a), ub = __float_as_uint(b);
  ua = (ua + 0x7FFFu + ((ua >> 16) & 1u)) >> 16;
  ub = (ub + 0x7FFFu + ((ub >> 16) & 1u)) >> 16;
  return ua | (ub << 16);
}
__device__ __forceinline__ unsigned short f2bf(float f) {
  unsigned u = __float_as_uint(f);
  u = (u + 0x7FFFu + ((u >> 16) & 1u)) >> 16;
  return (unsigned short)u;
}

// ---------------- init / convert -------------------------------------------
__global__ void init_kernel(int* count) {
  if (threadIdx.x < NEXP) count[threadIdx.x] = 0;
}

__global__ __launch_bounds__(256) void cvt_kernel(const float* __restrict__ x,
                                                  unsigned short* __restrict__ y) {
  int i = (blockIdx.x * 256 + threadIdx.x) * 4;
  float4 f = *(const float4*)(x + i);
  uint2 o;
  o.x = pk2(f.x, f.y);
  o.y = pk2(f.z, f.w);
  *(uint2*)(y + i) = o;
}

// ---------------- gate: softmax + top-2 + routing lists --------------------
__global__ __launch_bounds__(256) void gate_kernel(const float* __restrict__ x,
                                                   const float* __restrict__ Wg,
                                                   int* __restrict__ count,
                                                   int* __restrict__ list,
                                                   float* __restrict__ wlist) {
  __shared__ float sWg[NEXP * DDIM];
  for (int i = threadIdx.x; i < NEXP * DDIM; i += 256) sWg[i] = Wg[i];
  __syncthreads();
  const int lane = threadIdx.x & 31, wv = threadIdx.x >> 5;
  const int t = blockIdx.x * 8 + wv;
  float acc[NEXP];
#pragma unroll
  for (int e = 0; e < NEXP; ++e) acc[e] = 0.f;
  const float* xr = x + (size_t)t * DDIM;
  for (int d = lane; d < DDIM; d += 32) {
    float xv = xr[d];
#pragma unroll
    for (int e = 0; e < NEXP; ++e) acc[e] += xv * sWg[e * DDIM + d];
  }
#pragma unroll
  for (int off = 16; off > 0; off >>= 1)
#pragma unroll
    for (int e = 0; e < NEXP; ++e) acc[e] += __shfl_xor(acc[e], off, 32);
  if (lane == 0) {
    float mx = acc[0];
#pragma unroll
    for (int e = 1; e < NEXP; ++e) mx = fmaxf(mx, acc[e]);
    float p[NEXP], s = 0.f;
#pragma unroll
    for (int e = 0; e < NEXP; ++e) { p[e] = expf(acc[e] - mx); s += p[e]; }
    float inv = 1.f / s;
#pragma unroll
    for (int e = 0; e < NEXP; ++e) p[e] *= inv;
    int i1 = 0; float v1 = p[0];
#pragma unroll
    for (int e = 1; e < NEXP; ++e) if (p[e] > v1) { v1 = p[e]; i1 = e; }
    int i2 = -1; float v2 = -1.f;
#pragma unroll
    for (int e = 0; e < NEXP; ++e) if (e != i1 && p[e] > v2) { v2 = p[e]; i2 = e; }
    float wden = 1.f / (v1 + v2 + 1e-20f);
    int s1 = atomicAdd(&count[i1], 1);
    list[i1 * N_TOK + s1] = t; wlist[i1 * N_TOK + s1] = v1 * wden;
    int s2 = atomicAdd(&count[i2], 1);
    list[i2 * N_TOK + s2] = t; wlist[i2 * N_TOK + s2] = v2 * wden;
  }
}

__global__ void offsets_kernel(const int* __restrict__ count, int* __restrict__ offs) {
  if (threadIdx.x == 0 && blockIdx.x == 0) {
    int o = 0;
    for (int e = 0; e < NEXP; ++e) {
      offs[e] = o;
      o += (count[e] + 127) & ~127;
    }
    offs[NEXP] = o;
  }
}

// ---------------- gather selected token rows (bf16) ------------------------
__global__ __launch_bounds__(256) void gather_kernel(const unsigned short* __restrict__ xbf,
                                                     unsigned short* __restrict__ xg,
                                                     const int* __restrict__ count,
                                                     const int* __restrict__ offs,
                                                     const int* __restrict__ list) {
  const int e = blockIdx.z, slot = blockIdx.x;
  if (slot >= count[e]) return;
  const int t = list[e * N_TOK + slot];
  const uint2* src = (const uint2*)(xbf + (size_t)t * DDIM);
  uint2* dst = (uint2*)(xg + (size_t)(offs[e] + slot) * DDIM);
  dst[threadIdx.x] = src[threadIdx.x];  // 256 threads x 8B = 1024 bf16
}

// ---------------- WMMA GEMM: Y = epilogue(A[bf16] * W[f32]^T + b) ----------
// A tile (bf16, 128x64) is DMA'd straight into LDS with global_load_async_to_lds_b128,
// ping-ponged so the async copy for step i+1 overlaps the WMMA math of step i.
// W tile (fp32) is loaded to VGPRs, converted to bf16, stored to LDS (single buffer).
// sched_group_barrier pins the pipeline: front-load DS reads, then WMMAs back-to-back.
// MODE 0: ReLU + BN(eval) -> bf16 store   (layers 1,2)
// MODE 1: sigmoid -> plain store to out   (shared layer 3)
// MODE 2: sigmoid * gate weight -> atomicAdd scatter to out (routed layer 3)
template <int MODE>
__global__ __launch_bounds__(256) void gemm_kernel(
    const unsigned short* __restrict__ Abase, const float* __restrict__ Wbase,
    const float* __restrict__ bias, const float* __restrict__ bnG,
    const float* __restrict__ bnB, const float* __restrict__ bnM,
    const float* __restrict__ bnV, unsigned short* __restrict__ Hout,
    float* __restrict__ out, const int* __restrict__ count,
    const int* __restrict__ offs, const int* __restrict__ list,
    const float* __restrict__ wlist, int Mfixed, int Nc, int Kd) {
  const int e = blockIdx.z;
  int rowOff = 0, padM, cnt;
  const float* W = Wbase;
  const float* bi = bias;
  const float *g = bnG, *bb = bnB, *mm = bnM, *vv = bnV;
  if (count) {  // grouped (routed experts)
    rowOff = offs[e];
    padM = offs[e + 1] - rowOff;
    cnt = count[e];
    W += (size_t)e * Nc * Kd;
    bi += (size_t)e * Nc;
    if (MODE == 0) { g += e * Nc; bb += e * Nc; mm += e * Nc; vv += e * Nc; }
  } else {
    padM = Mfixed; cnt = Mfixed;
  }
  const int bm = blockIdx.y * 128;
  const int bn = blockIdx.x * 128;
  if (bm >= padM) return;  // block-uniform early exit
  const unsigned short* A = Abase + (size_t)rowOff * Kd;

  __shared__ __align__(16) unsigned short As[2][128 * LDSS];  // ping-pong A tiles
  __shared__ __align__(16) unsigned short Bs[128 * LDSS];

  const int tid = threadIdx.x;
  const int lane = tid & 31, wv = tid >> 5;
  const int lmod = lane & 15, lhi = lane >> 4;

  // ---- A tile async-DMA setup: 4 x b128 per thread cover 128 rows x 64 bf16
  unsigned ldsOffA[4];
  unsigned voffA[4];
#pragma unroll
  for (int i = 0; i < 4; ++i) {
    int idx = tid + 256 * i;
    int row = idx >> 3;     // 8 x 16B segments per 128B row
    int seg = idx & 7;
    ldsOffA[i] = (unsigned)(row * LDSS + seg * 8) * 2u;
    voffA[i] = (unsigned)(((bm + row) * Kd + seg * 8) * 2);  // byte offset vs A
  }
  const unsigned asBase0 = (unsigned)(uintptr_t)&As[0][0];
  const unsigned asBase1 = (unsigned)(uintptr_t)&As[1][0];

  auto issueA = [&](unsigned base) {
#pragma unroll
    for (int i = 0; i < 4; ++i) {
      asm volatile("global_load_async_to_lds_b128 %0, %1, %2"
                   :
                   : "v"(base + ldsOffA[i]), "v"(voffA[i]), "s"(A)
                   : "memory");
      voffA[i] += 128;  // advance 64 bf16 per K step
    }
  };

  // ---- B tile staging: thread covers one half-row (32 floats)
  const int ldRow = tid >> 1;
  const int ldHalf = (tid & 1) * 32;
  const float* wG = W + (size_t)(bn + ldRow) * Kd + ldHalf;
  unsigned short* bsD = &Bs[ldRow * LDSS + ldHalf];

  v8f C[8];
  v8f zero = {0.f, 0.f, 0.f, 0.f, 0.f, 0.f, 0.f, 0.f};
#pragma unroll
  for (int n = 0; n < 8; ++n) C[n] = zero;

  const unsigned short* aP0 = &As[0][(16 * wv + lmod) * LDSS];
  const unsigned short* aP1 = &As[1][(16 * wv + lmod) * LDSS];

  union FragU { uint4 q[2]; v16bf v; };

  const int nk = Kd >> 6;
  issueA(asBase0);  // prologue: DMA first A tile into buffer 0

  for (int it = 0; it < nk; ++it) {
    const int kt = it << 6;
    float4 f[8];
#pragma unroll
    for (int j = 0; j < 8; ++j) f[j] = *(const float4*)(wG + kt + j * 4);

    __syncthreads();  // all waves done reading previous LDS contents

    if (it + 1 < nk) issueA((it + 1) & 1 ? asBase1 : asBase0);

    uint4 pq[4];
#pragma unroll
    for (int j = 0; j < 4; ++j) {
      pq[j].x = pk2(f[2 * j].x, f[2 * j].y);
      pq[j].y = pk2(f[2 * j].z, f[2 * j].w);
      pq[j].z = pk2(f[2 * j + 1].x, f[2 * j + 1].y);
      pq[j].w = pk2(f[2 * j + 1].z, f[2 * j + 1].w);
      *(uint4*)(bsD + j * 8) = pq[j];
    }

    // wait for THIS tile's async DMA (leave the 4 just-issued ones in flight)
    if (it + 1 < nk) {
      asm volatile("s_wait_asynccnt 0x4" ::: "memory");
    } else {
      asm volatile("s_wait_asynccnt 0x0" ::: "memory");
    }
    __syncthreads();

    const unsigned short* aP = (it & 1) ? aP1 : aP0;
    FragU af0, af1;
    af0.q[0] = *(const uint4*)(aP + 8 * lhi);
    af0.q[1] = *(const uint4*)(aP + 8 * lhi + 16);
    af1.q[0] = *(const uint4*)(aP + 32 + 8 * lhi);
    af1.q[1] = *(const uint4*)(aP + 32 + 8 * lhi + 16);

#pragma unroll
    for (int ks = 0; ks < 2; ++ks) {
      // load ALL 8 B fragments into distinct registers first -> incremental waits
      FragU bf[8];
#pragma unroll
      for (int n = 0; n < 8; ++n) {
        const unsigned short* bp = &Bs[(n * 16 + lmod) * LDSS + 32 * ks + 8 * lhi];
        bf[n].q[0] = *(const uint4*)bp;
        bf[n].q[1] = *(const uint4*)(bp + 16);
      }
#pragma unroll
      for (int n = 0; n < 8; ++n) {
        C[n] = __builtin_amdgcn_wmma_f32_16x16x32_bf16(
            false, (ks ? af1 : af0).v, false, bf[n].v, (short)0, C[n], false, false);
      }
    }

    // Pin the schedule for this region: issue the DS-read bursts ahead of the
    // WMMA chains so dscnt waits are incremental and WMMAs go back-to-back.
    // 0x100 = DS read group, 0x8 = MFMA/WMMA group (best-effort directives).
    __builtin_amdgcn_sched_group_barrier(0x100, 20, 0);  // 4 A-frag + 16 B-frag (ks=0)
    __builtin_amdgcn_sched_group_barrier(0x008, 8, 0);   // 8 WMMA (ks=0)
    __builtin_amdgcn_sched_group_barrier(0x100, 16, 0);  // 16 B-frag (ks=1)
    __builtin_amdgcn_sched_group_barrier(0x008, 8, 0);   // 8 WMMA (ks=1)
  }

  // ---- epilogue; C layout: col = lane%16, row = r + 8*(lane/16)
  float bcA[8], scA[8], shA[8], mnA[8];
#pragma unroll
  for (int n = 0; n < 8; ++n) {
    const int col = bn + n * 16 + lmod;
    bcA[n] = bi[col];
    if (MODE == 0) {
      scA[n] = g[col] * rsqrtf(vv[col] + 1e-5f);
      shA[n] = bb[col];
      mnA[n] = mm[col];
    }
  }
  const int rowT = bm + 16 * wv + 8 * lhi;
#pragma unroll
  for (int r = 0; r < 8; ++r) {
    const int row = rowT + r;
    int tok = 0; float wgt = 0.f; bool act = true;
    if (MODE == 2) {
      act = row < cnt;
      if (act) {
        tok = list[e * N_TOK + row];
        wgt = wlist[e * N_TOK + row];
      }
    }
#pragma unroll
    for (int n = 0; n < 8; ++n) {
      const int col = bn + n * 16 + lmod;
      float val = C[n][r] + bcA[n];
      if (MODE == 0) {
        float t = fmaxf(val, 0.f);
        Hout[(size_t)(rowOff + row) * Nc + col] = f2bf(scA[n] * (t - mnA[n]) + shA[n]);
      } else {
        val = 1.f / (1.f + __expf(-val));
        if (MODE == 1) {
          out[(size_t)row * Nc + col] = val;  // shared expert writes every element
        } else if (act) {
          atomicAdd(&out[(size_t)tok * Nc + col], wgt * val);
        }
      }
    }
  }
}

// ---------------------------------------------------------------------------
extern "C" void kernel_launch(void* const* d_in, const int* in_sizes, int n_in,
                              void* d_out, int out_size, void* d_ws, size_t ws_size,
                              hipStream_t stream) {
  const float* x   = (const float*)d_in[0];
  const float* Wg  = (const float*)d_in[1];
  const float* W1  = (const float*)d_in[2];
  const float* b1  = (const float*)d_in[3];
  const float* g1  = (const float*)d_in[4];
  const float* be1 = (const float*)d_in[5];
  const float* m1  = (const float*)d_in[6];
  const float* v1  = (const float*)d_in[7];
  const float* W2  = (const float*)d_in[8];
  const float* b2  = (const float*)d_in[9];
  const float* g2  = (const float*)d_in[10];
  const float* be2 = (const float*)d_in[11];
  const float* m2  = (const float*)d_in[12];
  const float* v2  = (const float*)d_in[13];
  const float* W3  = (const float*)d_in[14];
  const float* b3  = (const float*)d_in[15];
  const float* sW1 = (const float*)d_in[16];
  const float* sb1 = (const float*)d_in[17];
  const float* sg1 = (const float*)d_in[18];
  const float* sbe1= (const float*)d_in[19];
  const float* sm1 = (const float*)d_in[20];
  const float* sv1 = (const float*)d_in[21];
  const float* sW2 = (const float*)d_in[22];
  const float* sb2 = (const float*)d_in[23];
  const float* sg2 = (const float*)d_in[24];
  const float* sbe2= (const float*)d_in[25];
  const float* sm2 = (const float*)d_in[26];
  const float* sv2 = (const float*)d_in[27];
  const float* sW3 = (const float*)d_in[28];
  const float* sb3 = (const float*)d_in[29];
  float* out = (float*)d_out;
  (void)in_sizes; (void)n_in; (void)out_size; (void)ws_size;

  // workspace layout (regions aliased across phases; stream order serializes)
  char* ws = (char*)d_ws;
  size_t off = 0;
  auto take = [&](size_t bytes) {
    char* p = ws + off;
    off = (off + bytes + 255) & ~(size_t)255;
    return p;
  };
  const size_t CAP_ROWS = 17408;  // sum of per-expert 128-padded counts (worst case)
  int* count = (int*)take(NEXP * 4);
  int* offs  = (int*)take((NEXP + 1) * 4);
  int* list  = (int*)take((size_t)NEXP * N_TOK * 4);
  float* wl  = (float*)take((size_t)NEXP * N_TOK * 4);
  unsigned short* xbf = (unsigned short*)take((size_t)N_TOK * DDIM * 2);
  size_t szXg = CAP_ROWS * (size_t)DDIM * 2;
  size_t szS  = (size_t)N_TOK * HDIM * 2;
  unsigned short* regA = (unsigned short*)take(szXg > szS ? szXg : szS);  // s1 | xg
  unsigned short* regB = (unsigned short*)take(CAP_ROWS * (size_t)HDIM * 2);  // s2 | h1
  unsigned short* regC = (unsigned short*)take(CAP_ROWS * (size_t)HDIM * 2);  // h2

  init_kernel<<<1, 32, 0, stream>>>(count);
  cvt_kernel<<<(N_TOK * DDIM) / 1024, 256, 0, stream>>>(x, xbf);
  gate_kernel<<<N_TOK / 8, 256, 0, stream>>>(x, Wg, count, list, wl);
  offsets_kernel<<<1, 1, 0, stream>>>(count, offs);

  // shared expert: x -> s1 -> s2 -> out (plain stores cover all elements)
  gemm_kernel<0><<<dim3(HDIM / 128, N_TOK / 128, 1), 256, 0, stream>>>(
      xbf, sW1, sb1, sg1, sbe1, sm1, sv1, regA, nullptr,
      nullptr, nullptr, nullptr, nullptr, N_TOK, HDIM, DDIM);
  gemm_kernel<0><<<dim3(HDIM / 128, N_TOK / 128, 1), 256, 0, stream>>>(
      regA, sW2, sb2, sg2, sbe2, sm2, sv2, regB, nullptr,
      nullptr, nullptr, nullptr, nullptr, N_TOK, HDIM, HDIM);
  gemm_kernel<1><<<dim3(ODIM / 128, N_TOK / 128, 1), 256, 0, stream>>>(
      regB, sW3, sb3, nullptr, nullptr, nullptr, nullptr, nullptr, out,
      nullptr, nullptr, nullptr, nullptr, N_TOK, ODIM, HDIM);

  // routed experts (top-2 only, grouped GEMM over compacted slot space)
  gather_kernel<<<dim3(N_TOK, 1, NEXP), 256, 0, stream>>>(xbf, regA, count, offs, list);
  gemm_kernel<0><<<dim3(HDIM / 128, N_TOK / 128, NEXP), 256, 0, stream>>>(
      regA, W1, b1, g1, be1, m1, v1, regB, nullptr,
      count, offs, nullptr, nullptr, 0, HDIM, DDIM);
  gemm_kernel<0><<<dim3(HDIM / 128, N_TOK / 128, NEXP), 256, 0, stream>>>(
      regB, W2, b2, g2, be2, m2, v2, regC, nullptr,
      count, offs, nullptr, nullptr, 0, HDIM, HDIM);
  gemm_kernel<2><<<dim3(ODIM / 128, N_TOK / 128, NEXP), 256, 0, stream>>>(
      regC, W3, b3, nullptr, nullptr, nullptr, nullptr, nullptr, out,
      count, offs, list, wl, 0, ODIM, HDIM);
}
